// SequenceGatedRecurrence_33990371181326
// MI455X (gfx1250) — compile-verified, matrix operand order checked by reading, not run on hardware
//
#include <hip/hip_runtime.h>
#include <hip/hip_bf16.h>
#include <math.h>

// ---------------------------------------------------------------------------
// Fused GRU-like scan for MI455X (gfx1250, wave32, WMMA).
//   B=64, T=2048, D=512, K=2D=1024.
// 16 workgroups x 256 threads (8 waves). Wave (g,w) owns the 16x16 hidden tile
//   rows [16r,16r+16), cols [32g+16ct, +16); 128 waves cover hidden[64,512],
//   accumulating in registers across all 2048 steps.
// Weights: bf16, WMMA-B-fragment swizzled, resident in 128 KB LDS per WG.
// A operand ("joined" = [hidden | x_t], bf16, A-fragment swizzled): ping-pong
//   pair of 128 KB buffers in workspace (L2-resident) -> 1 grid barrier/step.
// 4 independent WMMA accumulator chains hide the D->C hazard window; biases
//   are folded into accumulator init; activations are branch-free v_exp/v_rcp.
// ---------------------------------------------------------------------------

typedef __bf16 bf16_t;
typedef __attribute__((ext_vector_type(16))) __bf16 v16bf;
typedef __attribute__((ext_vector_type(8)))  float  v8f;
typedef __attribute__((ext_vector_type(2)))  __bf16 bf16x2;

#define NB 64
#define NT 2048
#define ND 512
#define NWG 16
#define NTHREADS 256
#define JOINED_ELEMS (64 * 1024)            // bf16 elems per buffer (128 KB)
#define JOINED_BYTES (JOINED_ELEMS * 2)
#define LDS_BYTES    (2 * 2 * 32 * 32 * 16 * 2) // 2 mats x 2 ct x 32 kch x 32 ln x 16 halves

// K index (within a 32-wide K chunk) held by half-lane group `hi` at half `i`
// of a 16x32 bf16 A fragment (ISA 7.12.2); B fragment mirrors with lane->N.
__device__ __forceinline__ int kmap(int hi, int i) {
  int v = i >> 1, pos = i & 1;
  int k = (v < 4) ? (2 * v + pos) : (16 + 2 * (v - 4) + pos);
  return k + (hi ? 8 : 0);
}

__device__ __forceinline__ void grid_barrier(unsigned* cnt, unsigned* gen) {
  __builtin_amdgcn_fence(__ATOMIC_RELEASE, "agent");
  __syncthreads();
  if (threadIdx.x == 0) {
    unsigned g = __hip_atomic_load(gen, __ATOMIC_RELAXED, __HIP_MEMORY_SCOPE_AGENT);
    unsigned a = __hip_atomic_fetch_add(cnt, 1u, __ATOMIC_ACQ_REL, __HIP_MEMORY_SCOPE_AGENT);
    if (a == (unsigned)(NWG - 1)) {
      __hip_atomic_store(cnt, 0u, __ATOMIC_RELAXED, __HIP_MEMORY_SCOPE_AGENT);
      __hip_atomic_store(gen, g + 1u, __ATOMIC_RELEASE, __HIP_MEMORY_SCOPE_AGENT);
    } else {
      while (__hip_atomic_load(gen, __ATOMIC_ACQUIRE, __HIP_MEMORY_SCOPE_AGENT) == g) {
        __builtin_amdgcn_s_sleep(1);
      }
    }
  }
  __syncthreads();
  __builtin_amdgcn_fence(__ATOMIC_ACQUIRE, "agent");
}

extern "C" __global__ void __launch_bounds__(NTHREADS)
gru_fused_kernel(const float* __restrict__ seq,
                 const float* __restrict__ Wg, const float* __restrict__ bg,
                 const float* __restrict__ Wc, const float* __restrict__ bc,
                 float* __restrict__ out,
                 bf16_t* __restrict__ joined,   // 2 x JOINED_ELEMS (ping-pong)
                 unsigned* cnt, unsigned* gen)
{
  extern __shared__ char ldsRaw[];
  bf16_t* ldsW = (bf16_t*)ldsRaw;

  const int tid  = threadIdx.x;
  const int wv   = tid >> 5;
  const int lane = tid & 31;
  const int hi   = lane >> 4;
  const int g    = blockIdx.x;
  const int ct   = wv >> 2;                 // column sub-tile (0..1)
  const int r    = wv & 3;                  // row tile (0..3)
  const int colbase = g * 32 + ct * 16;
  const int col  = colbase + (lane & 15);   // this lane's N column (fixed)

  // ---- one-time: convert + swizzle weights into LDS (B-fragment layout) ----
  for (int mat = 0; mat < 2; ++mat) {
    const float* W = mat ? Wc : Wg;
    for (int kch = r; kch < 32; kch += 4) {
      size_t lbase = ((((size_t)mat * 2 + ct) * 32 + kch) * 32 + lane) * 16;
      #pragma unroll
      for (int i = 0; i < 16; ++i) {
        int K = kch * 32 + kmap(hi, i);
        ldsW[lbase + i] = (bf16_t)W[(size_t)K * ND + col];
      }
    }
  }

  const float bgv = bg[col];
  const float bcv = bc[col];

  float h[8];
  #pragma unroll
  for (int j = 0; j < 8; ++j) h[j] = 0.f;

  // scatter coords for writing bf16 hidden into the A-fragment buffer:
  // hidden column `col` is joined-K index `col`.
  const int k_in    = col & 31;
  const int kchH    = col >> 5;
  const int vH      = ((k_in & 16) ? 4 : 0) + ((k_in & 7) >> 1);
  const int iH      = 2 * vH + (k_in & 1);
  const int laneHiH = (k_in & 8) ? 16 : 0;

  // x_t staging: 64 frag-blocks (4 row tiles x 16 K-chunks) over first 64 waves
  const int Wid = g * 8 + wv;
  auto stage_x = [&](int t, bf16_t* dst) {
    if (Wid < 64) {
      const int rs = Wid >> 4, kx = Wid & 15;
      const int m  = rs * 16 + (lane & 15);
      const float* xrow = seq + ((size_t)m * NT + t) * ND + kx * 32;
      const size_t fbase = (((size_t)rs * 32 + (16 + kx)) * 32 + lane) * 16;
      #pragma unroll
      for (int v = 0; v < 8; ++v) {
        int k0 = kmap(hi, 2 * v);                 // even; k0,k0+1 consecutive
        float2 xv = *(const float2*)(xrow + k0);
        bf16x2 p; p.x = (bf16_t)xv.x; p.y = (bf16_t)xv.y;
        *(bf16x2*)(dst + fbase + 2 * v) = p;
      }
    }
  };

  stage_x(0, joined);          // hidden half of buffer 0 is zero via memset
  grid_barrier(cnt, gen);

  const bf16_t* bgBase = ldsW + (((size_t)0 * 2 + ct) * 32) * 32 * 16;
  const bf16_t* bcBase = ldsW + (((size_t)1 * 2 + ct) * 32) * 32 * 16;

  for (int t = 0; t < NT; ++t) {
    const bf16_t* rdBuf = joined + (size_t)(t & 1) * JOINED_ELEMS;
    bf16_t*       wrBuf = (bf16_t*)joined + (size_t)((t + 1) & 1) * JOINED_ELEMS;
    const bf16_t* aBase = rdBuf + (size_t)r * 32 * 32 * 16;

    // 4 independent accumulator chains; biases folded into chain 0 init.
    v8f ag0, ag1, ac0, ac1;
    #pragma unroll
    for (int j = 0; j < 8; ++j) { ag0[j] = bgv; ac0[j] = bcv; ag1[j] = 0.f; ac1[j] = 0.f; }

    #pragma unroll 4
    for (int k = 0; k < 32; k += 2) {
      v16bf a0  = *(const v16bf*)(aBase  + ((size_t)(k    ) * 32 + lane) * 16);
      v16bf Bg0 = *(const v16bf*)(bgBase + ((size_t)(k    ) * 32 + lane) * 16);
      v16bf Bc0 = *(const v16bf*)(bcBase + ((size_t)(k    ) * 32 + lane) * 16);
      v16bf a1  = *(const v16bf*)(aBase  + ((size_t)(k + 1) * 32 + lane) * 16);
      v16bf Bg1 = *(const v16bf*)(bgBase + ((size_t)(k + 1) * 32 + lane) * 16);
      v16bf Bc1 = *(const v16bf*)(bcBase + ((size_t)(k + 1) * 32 + lane) * 16);
      ag0 = __builtin_amdgcn_wmma_f32_16x16x32_bf16(false, a0, false, Bg0,
                                                    (short)0, ag0, false, false);
      ac0 = __builtin_amdgcn_wmma_f32_16x16x32_bf16(false, a0, false, Bc0,
                                                    (short)0, ac0, false, false);
      ag1 = __builtin_amdgcn_wmma_f32_16x16x32_bf16(false, a1, false, Bg1,
                                                    (short)0, ag1, false, false);
      ac1 = __builtin_amdgcn_wmma_f32_16x16x32_bf16(false, a1, false, Bc1,
                                                    (short)0, ac1, false, false);
    }

    // Branch-free activations: sigma(x)=rcp(1+exp(-x)), tanh(x)=2*sigma(2x)-1.
    #pragma unroll
    for (int j = 0; j < 8; ++j) {
      float sg = ag0[j] + ag1[j];
      float sc = ac0[j] + ac1[j];
      float gate = __builtin_amdgcn_rcpf(1.f + __expf(-sg));
      float th   = 2.f * __builtin_amdgcn_rcpf(1.f + __expf(-2.f * sc)) - 1.f;
      h[j] = gate * (h[j] - th) + th;       // == gate*h + (1-gate)*tanh
    }

    if (t + 1 < NT) {
      #pragma unroll
      for (int j = 0; j < 8; ++j) {          // scatter new hidden tile (bf16)
        int mloc  = j + 8 * hi;
        int lanep = mloc + laneHiH;
        wrBuf[(((size_t)r * 32 + kchH) * 32 + lanep) * 16 + iH] = (bf16_t)h[j];
      }
      stage_x(t + 1, wrBuf);
      grid_barrier(cnt, gen);  // all writes to wrBuf done before anyone reads
    }
  }

  #pragma unroll
  for (int j = 0; j < 8; ++j) {
    int m = 16 * r + j + 8 * hi;
    out[(size_t)m * ND + col] = h[j];
  }
}

extern "C" void kernel_launch(void* const* d_in, const int* in_sizes, int n_in,
                              void* d_out, int out_size, void* d_ws, size_t ws_size,
                              hipStream_t stream) {
  const float* seq = (const float*)d_in[0];
  const float* Wg  = (const float*)d_in[1];
  const float* bgv = (const float*)d_in[2];
  const float* Wc  = (const float*)d_in[3];
  const float* bcv = (const float*)d_in[4];
  float* out = (float*)d_out;

  unsigned* cnt  = (unsigned*)d_ws;
  unsigned* gen  = (unsigned*)((char*)d_ws + 256);
  bf16_t* joined = (bf16_t*)((char*)d_ws + 1024);   // 2 ping-pong buffers

  // Zero barrier state + both joined buffers (hidden of buffer 0 must be 0).
  hipMemsetAsync(d_ws, 0, 1024 + 2 * (size_t)JOINED_BYTES, stream);

  // Allow 128 KB dynamic LDS per workgroup (320 KB available per WGP).
  hipFuncSetAttribute((const void*)gru_fused_kernel,
                      hipFuncAttributeMaxDynamicSharedMemorySize, LDS_BYTES);

  dim3 grid(NWG), block(NTHREADS);
  hipLaunchKernelGGL(gru_fused_kernel, grid, block, LDS_BYTES, stream,
                     seq, Wg, bgv, Wc, bcv, out, joined, cnt, gen);
}
